// MultiHeadAttention_12386685681783
// MI455X (gfx1250) — matrix operand order repaired and checked by reading, not compiled, not run on hardware
//
#include <hip/hip_runtime.h>
#include <hip/hip_bf16.h>

typedef __attribute__((ext_vector_type(16))) __bf16 v16bf;
typedef __attribute__((ext_vector_type(8)))  __bf16 v8bf;
typedef __attribute__((ext_vector_type(8)))  float  v8f;

#define EMB   1024
#define SEQ   2048
#define NHEAD 16
#define DK    64
#define MTOT  4096   // B*S

static __device__ __forceinline__ v16bf frag16(const __bf16* p0, const __bf16* p1) {
  v8bf lo = *(const v8bf*)p0;
  v8bf hi = *(const v8bf*)p1;
  v16bf r;
#pragma unroll
  for (int i = 0; i < 8; i++) { r[i] = lo[i]; r[i + 8] = hi[i]; }
  return r;
}

__global__ void cvt_f32_bf16(const float* __restrict__ src, __bf16* __restrict__ dst, int n) {
  int i = blockIdx.x * blockDim.x + threadIdx.x;
  int stride = gridDim.x * blockDim.x;
  for (; i < n; i += stride) dst[i] = (__bf16)src[i];
}

// C = (A(MxK=1024) * W(N=1024 x K=1024)^T + bias) * oscale, bf16 in, f32 accumulate.
// mode 0: out bf16 head-split [B,H,S,64]
// mode 1: out bf16 head-split transposed [B,H,64,S]
// mode 2: out f32 row-major [M,1024] (final projection -> d_out)
__global__ __launch_bounds__(256)
void gemm_bf16_k1024(const __bf16* __restrict__ A, const __bf16* __restrict__ W,
                     const float* __restrict__ bias, __bf16* __restrict__ outb,
                     float* __restrict__ outf, int mode, float oscale)
{
  __shared__ __align__(16) __bf16 As[128 * 32];
  __shared__ __align__(16) __bf16 Bs[128 * 32];   // stored [n][k] (column-of-B contiguous in K)
  const int m0   = blockIdx.x * 128;
  const int n0   = blockIdx.y * 128;
  const int t    = threadIdx.x;
  const int lane = t & 31;
  const int w    = t >> 5;
  const int wm   = w >> 2;        // 0..1  -> 64 rows each
  const int wn   = w & 3;         // 0..3  -> 32 cols each
  const int half = lane >> 4;
  const int l16  = lane & 15;
  const int lrow = t >> 1;        // 0..127
  const int lcol = (t & 1) * 16;  // 0 / 16

  v8f acc[4][2];
  const v8f vzero = {0.f, 0.f, 0.f, 0.f, 0.f, 0.f, 0.f, 0.f};
#pragma unroll
  for (int i = 0; i < 4; i++)
#pragma unroll
    for (int j = 0; j < 2; j++) acc[i][j] = vzero;

  for (int k0 = 0; k0 < 1024; k0 += 32) {
    __syncthreads();
    {
      const __bf16* sa = A + (size_t)(m0 + lrow) * 1024 + k0 + lcol;
      *(v8bf*)(&As[lrow * 32 + lcol])     = *(const v8bf*)(sa);
      *(v8bf*)(&As[lrow * 32 + lcol + 8]) = *(const v8bf*)(sa + 8);
      const __bf16* sb = W + (size_t)(n0 + lrow) * 1024 + k0 + lcol;
      *(v8bf*)(&Bs[lrow * 32 + lcol])     = *(const v8bf*)(sb);
      *(v8bf*)(&Bs[lrow * 32 + lcol + 8]) = *(const v8bf*)(sb + 8);
    }
    __syncthreads();

    v16bf af[4], bf[2];
#pragma unroll
    for (int i = 0; i < 4; i++) {
      const int row = wm * 64 + i * 16 + l16;
      const __bf16* p = &As[row * 32 + half * 8];
      af[i] = frag16(p, p + 16);
    }
#pragma unroll
    for (int j = 0; j < 2; j++) {
      const int col = wn * 32 + j * 16 + l16;
      const __bf16* p = &Bs[col * 32 + half * 8];
      bf[j] = frag16(p, p + 16);
    }
#pragma unroll
    for (int i = 0; i < 4; i++)
#pragma unroll
      for (int j = 0; j < 2; j++)
        acc[i][j] = __builtin_amdgcn_wmma_f32_16x16x32_bf16(
            false, af[i], false, bf[j], (short)0, acc[i][j], false, false);
  }

#pragma unroll
  for (int i = 0; i < 4; i++) {
#pragma unroll
    for (int j = 0; j < 2; j++) {
      const int n = n0 + wn * 32 + j * 16 + l16;
      const float bv = bias[n];
#pragma unroll
      for (int r = 0; r < 8; r++) {
        const int m = m0 + wm * 64 + i * 16 + r + 8 * half;
        const float val = (acc[i][j][r] + bv) * oscale;
        if (mode == 2) {
          outf[(size_t)m * 1024 + n] = val;
        } else {
          const int b = m >> 11, s = m & 2047;
          const int h = n >> 6, d = n & 63;
          size_t idx;
          if (mode == 0) idx = ((size_t)(b * NHEAD + h) * SEQ + s) * DK + d;
          else           idx = ((size_t)(b * NHEAD + h) * DK + d) * SEQ + s;
          outb[idx] = (__bf16)val;
        }
      }
    }
  }
}

// Flash attention: each wave owns 16 query rows of one (b,h); 64 K-columns/iter.
// Qh (pre-scaled by 1/sqrt(dk)), Kh: [B*H, S, 64] bf16.  VT: [B*H, 64, S] bf16.
// X: [B, S, E] bf16.
__global__ __launch_bounds__(256)
void flash_attn(const __bf16* __restrict__ Qh, const __bf16* __restrict__ Kh,
                const __bf16* __restrict__ VT, __bf16* __restrict__ X)
{
  __shared__ __align__(16) __bf16 Plds[8 * 16 * 64];   // per-wave 16x64 P tile
  const int t    = threadIdx.x;
  const int lane = t & 31;
  const int w    = t >> 5;
  const int half = lane >> 4;
  const int l16  = lane & 15;
  const int bh   = blockIdx.x >> 4;   // 0..31
  const int qblk = blockIdx.x & 15;
  const int q0   = qblk * 128 + w * 16;

  const __bf16* Qb = Qh + (size_t)bh * SEQ * DK;
  const __bf16* Kb = Kh + (size_t)bh * SEQ * DK;
  const __bf16* Vb = VT + (size_t)bh * DK * SEQ;

  v16bf aq[2];
  {
    const __bf16* qr = Qb + (size_t)(q0 + l16) * DK;
#pragma unroll
    for (int c = 0; c < 2; c++) {
      const __bf16* p = qr + c * 32 + half * 8;
      aq[c] = frag16(p, p + 16);
    }
  }

  float mrow[8], lrow[8];
  v8f oacc[4];
  const v8f vzero = {0.f, 0.f, 0.f, 0.f, 0.f, 0.f, 0.f, 0.f};
#pragma unroll
  for (int r = 0; r < 8; r++) { mrow[r] = -1e30f; lrow[r] = 0.f; }
#pragma unroll
  for (int n = 0; n < 4; n++) oacc[n] = vzero;

  __bf16* myP = &Plds[w * (16 * 64)];

  for (int kb = 0; kb < SEQ; kb += 64) {
    // scores: four 16x16 tiles covering 64 K-columns
    v8f st[4];
#pragma unroll
    for (int ti = 0; ti < 4; ti++) {
      v8f s = vzero;
      const __bf16* kr = Kb + (size_t)(kb + ti * 16 + l16) * DK;
#pragma unroll
      for (int c = 0; c < 2; c++) {
        const __bf16* p = kr + c * 32 + half * 8;
        v16bf bk = frag16(p, p + 16);
        s = __builtin_amdgcn_wmma_f32_16x16x32_bf16(
            false, aq[c], false, bk, (short)0, s, false, false);
      }
      st[ti] = s;
    }

    // online softmax over 64 columns (Q already carries the 1/sqrt(dk) scale)
    float pe[4][8];
#pragma unroll
    for (int r = 0; r < 8; r++) {
      float mx = fmaxf(fmaxf(st[0][r], st[1][r]), fmaxf(st[2][r], st[3][r]));
      mx = fmaxf(mx, __shfl_xor(mx, 1, 32));
      mx = fmaxf(mx, __shfl_xor(mx, 2, 32));
      mx = fmaxf(mx, __shfl_xor(mx, 4, 32));
      mx = fmaxf(mx, __shfl_xor(mx, 8, 32));
      const float mnew = fmaxf(mrow[r], mx);
      const float fac  = __expf(mrow[r] - mnew);
      float rs = 0.f;
#pragma unroll
      for (int ti = 0; ti < 4; ti++) {
        const float e = __expf(st[ti][r] - mnew);
        pe[ti][r] = e;
        rs += e;
      }
      rs += __shfl_xor(rs, 1, 32);
      rs += __shfl_xor(rs, 2, 32);
      rs += __shfl_xor(rs, 4, 32);
      rs += __shfl_xor(rs, 8, 32);
      lrow[r] = lrow[r] * fac + rs;
      mrow[r] = mnew;
#pragma unroll
      for (int n = 0; n < 4; n++) oacc[n][r] = oacc[n][r] * fac;
    }

    // C-fragment -> A-fragment reshape via per-wave LDS tile [16][64]
#pragma unroll
    for (int r = 0; r < 8; r++) {
      const int row = r + 8 * half;
#pragma unroll
      for (int ti = 0; ti < 4; ti++)
        myP[row * 64 + ti * 16 + l16] = (__bf16)pe[ti][r];
    }
    v16bf ap[2];
#pragma unroll
    for (int c = 0; c < 2; c++) {
      const __bf16* p = &myP[l16 * 64 + c * 32 + half * 8];
      ap[c] = frag16(p, p + 16);
    }

    // P @ V over the 64-k block
#pragma unroll
    for (int n = 0; n < 4; n++) {
#pragma unroll
      for (int c = 0; c < 2; c++) {
        const __bf16* vr = Vb + (size_t)(n * 16 + l16) * SEQ + kb + c * 32 + half * 8;
        v16bf bv = frag16(vr, vr + 16);
        oacc[n] = __builtin_amdgcn_wmma_f32_16x16x32_bf16(
            false, ap[c], false, bv, (short)0, oacc[n], false, false);
      }
    }
  }

  const int b = bh >> 4;
  const int h = bh & 15;
#pragma unroll
  for (int n = 0; n < 4; n++) {
#pragma unroll
    for (int r = 0; r < 8; r++) {
      const int s = q0 + r + 8 * half;
      const int d = n * 16 + l16;
      const float val = oacc[n][r] / lrow[r];
      X[((size_t)(b * SEQ + s)) * EMB + h * DK + d] = (__bf16)val;
    }
  }
}

extern "C" void kernel_launch(void* const* d_in, const int* in_sizes, int n_in,
                              void* d_out, int out_size, void* d_ws, size_t ws_size,
                              hipStream_t stream) {
  (void)in_sizes; (void)n_in; (void)out_size; (void)ws_size;
  const float* query = (const float*)d_in[0];
  const float* key   = (const float*)d_in[1];
  const float* value = (const float*)d_in[2];
  const float* Wq = (const float*)d_in[3];
  const float* bq = (const float*)d_in[4];
  const float* Wk = (const float*)d_in[5];
  const float* bk = (const float*)d_in[6];
  const float* Wv = (const float*)d_in[7];
  const float* bv = (const float*)d_in[8];
  const float* Wo = (const float*)d_in[9];
  const float* bo = (const float*)d_in[10];
  float* out = (float*)d_out;

  __bf16* ws = (__bf16*)d_ws;
  const size_t MK = (size_t)MTOT * EMB;  // 4M elems
  const size_t WW = (size_t)EMB * EMB;   // 1M elems
  __bf16* qx = ws;
  __bf16* kx = qx + MK;
  __bf16* vx = kx + MK;
  __bf16* wq = vx + MK;
  __bf16* wk = wq + WW;
  __bf16* wv = wk + WW;
  __bf16* wo = wv + WW;
  __bf16* Qh = wo + WW;
  __bf16* Kh = Qh + MK;
  __bf16* VT = Kh + MK;
  __bf16* X  = VT + MK;

  const int cvtBlocks = 2048;  // grid-stride
  cvt_f32_bf16<<<cvtBlocks, 256, 0, stream>>>(query, qx, (int)MK);
  cvt_f32_bf16<<<cvtBlocks, 256, 0, stream>>>(key,   kx, (int)MK);
  cvt_f32_bf16<<<cvtBlocks, 256, 0, stream>>>(value, vx, (int)MK);
  cvt_f32_bf16<<<cvtBlocks, 256, 0, stream>>>(Wq, wq, (int)WW);
  cvt_f32_bf16<<<cvtBlocks, 256, 0, stream>>>(Wk, wk, (int)WW);
  cvt_f32_bf16<<<cvtBlocks, 256, 0, stream>>>(Wv, wv, (int)WW);
  cvt_f32_bf16<<<cvtBlocks, 256, 0, stream>>>(Wo, wo, (int)WW);

  dim3 gg(MTOT / 128, EMB / 128);
  // Fold 1/sqrt(dk)=0.125 into the Q projection (linear, so equivalent).
  gemm_bf16_k1024<<<gg, 256, 0, stream>>>(qx, wq, bq, Qh, nullptr, 0, 0.125f);
  gemm_bf16_k1024<<<gg, 256, 0, stream>>>(kx, wk, bk, Kh, nullptr, 0, 1.0f);
  gemm_bf16_k1024<<<gg, 256, 0, stream>>>(vx, wv, bv, VT, nullptr, 1, 1.0f);

  flash_attn<<<512, 256, 0, stream>>>(Qh, Kh, VT, X);

  gemm_bf16_k1024<<<gg, 256, 0, stream>>>(X, wo, bo, nullptr, out, 2, 1.0f);
}
